// SeedGenerator_90623809946175
// MI455X (gfx1250) — matrix operand order repaired and checked by reading, not compiled, not run on hardware
//
#include <hip/hip_runtime.h>
#include <hip/hip_bf16.h>

// ---------------------------------------------------------------------------
// SeedGenerator for MI455X (gfx1250, wave32, WMMA).
//  Stage 0: f32 -> f16 conversion of feat + all MLP weights (workspace).
//  Stage 1: x1 = feat(64x512) @ ps_w(512x65536) + ps_b via v_wmma_f32_16x16x32_f16,
//           ps_w double-buffered through LDS (read from HBM exactly once),
//           A-fragments prefetched ahead of the barrier.
//  Stage 2: fully fused MLP-res stack, one WG per (batch, 32-point tile),
//           all activations resident in LDS, dual-accumulator pipelined WMMA.
// ---------------------------------------------------------------------------

typedef __attribute__((ext_vector_type(16))) _Float16 v16h;
typedef __attribute__((ext_vector_type(8)))  _Float16 v8h;
typedef __attribute__((ext_vector_type(8)))  float    v8f;

__device__ __forceinline__ v16h mk16(v8h lo, v8h hi) {
  return __builtin_shufflevector(lo, hi, 0,1,2,3,4,5,6,7,8,9,10,11,12,13,14,15);
}

__device__ __forceinline__ v8f wmma_f16(v16h a, v16h b, v8f c) {
  return __builtin_amdgcn_wmma_f32_16x16x32_f16(
      /*neg_a=*/false, a, /*neg_b=*/false, b,
      /*c_mod=*/(short)0, c, /*reuse_a=*/false, /*reuse_b=*/false);
}

// A fragment (16x32, M x K) from row-major f16 weights W[ld] at (row0, k0).
// Lane: M = row0 + (lane&15); K groups {half*8 + t, 16 + half*8 + t}.
__device__ __forceinline__ v16h loadA_g(const _Float16* __restrict__ W, int ld,
                                        int row0, int k0, int lane) {
  int m = row0 + (lane & 15);
  int h = lane >> 4;
  const _Float16* p = W + (size_t)m * ld + k0 + h * 8;
  v8h lo = *(const v8h*)p;
  v8h hi = *(const v8h*)(p + 16);
  return mk16(lo, hi);
}

// Guarded variant for the 3-row final weight matrix.
__device__ __forceinline__ v16h loadA_g_guard(const _Float16* __restrict__ W, int ld,
                                              int rows, int row0, int k0, int lane) {
  int m = row0 + (lane & 15);
  int h = lane >> 4;
  v8h z = {};
  v8h lo = z, hi = z;
  if (m < rows) {
    const _Float16* p = W + (size_t)m * ld + k0 + h * 8;
    lo = *(const v8h*)p;
    hi = *(const v8h*)(p + 16);
  }
  return mk16(lo, hi);
}

// B fragment (32x16, K x N) from LDS activations stored POINT-MAJOR:
// act[n][ch] with row stride ld halves, so B[k][n] = act[n][k] and each lane
// reads K groups {half*8 + t, 16 + half*8 + t} (two 16-byte chunks).
__device__ __forceinline__ v16h loadB_lds(const _Float16* act, int ld,
                                          int col0, int k0, int lane) {
  int n = col0 + (lane & 15);
  int h = lane >> 4;
  const _Float16* p = act + (size_t)n * ld + k0 + h * 8;
  v8h lo = *(const v8h*)p;
  v8h hi = *(const v8h*)(p + 16);
  return mk16(lo, hi);
}

// Single-tile pipelined GEMM: acc += W[row0:+16] x act[col0:+16].
__device__ __forceinline__ void gemm_acc(v8f& acc,
                                         const _Float16* __restrict__ W, int ldw, int row0,
                                         const _Float16* act, int ldact, int col0,
                                         int K, int lane) {
  v16h a = loadA_g(W, ldw, row0, 0, lane);
  v16h b = loadB_lds(act, ldact, col0, 0, lane);
  for (int k0 = 32; k0 < K; k0 += 32) {
    v16h an = loadA_g(W, ldw, row0, k0, lane);
    v16h bn = loadB_lds(act, ldact, col0, k0, lane);
    acc = wmma_f16(a, b, acc);
    a = an; b = bn;
  }
  acc = wmma_f16(a, b, acc);
}

// Dual-column pipelined GEMM: one A-fragment feeds two WMMAs (col tiles 0,16).
__device__ __forceinline__ void gemm_acc2(v8f& acc0, v8f& acc1,
                                          const _Float16* __restrict__ W, int ldw, int row0,
                                          const _Float16* act, int ldact,
                                          int K, int lane) {
  v16h a  = loadA_g(W, ldw, row0, 0, lane);
  v16h b0 = loadB_lds(act, ldact, 0, 0, lane);
  v16h b1 = loadB_lds(act, ldact, 16, 0, lane);
  for (int k0 = 32; k0 < K; k0 += 32) {
    v16h an  = loadA_g(W, ldw, row0, k0, lane);
    v16h b0n = loadB_lds(act, ldact, 0, k0, lane);
    v16h b1n = loadB_lds(act, ldact, 16, k0, lane);
    acc0 = wmma_f16(a, b0, acc0);
    acc1 = wmma_f16(a, b1, acc1);
    a = an; b0 = b0n; b1 = b1n;
  }
  acc0 = wmma_f16(a, b0, acc0);
  acc1 = wmma_f16(a, b1, acc1);
}

// C tile (16x16 f32) -> LDS act[n][m] f16, +bias (+ReLU). Lane holds 8
// consecutive M values => single v8h store.
__device__ __forceinline__ void store_act_relu(const v8f& acc, const float* __restrict__ bias,
                                               _Float16* act, int ld,
                                               int row0, int col0, int lane) {
  int n  = col0 + (lane & 15);
  int mb = row0 + ((lane >> 4) << 3);
  v8h o;
#pragma unroll
  for (int r = 0; r < 8; r++) {
    float v = acc[r] + bias[mb + r];
    o[r] = (_Float16)(v > 0.f ? v : 0.f);
  }
  *(v8h*)(act + (size_t)n * ld + mb) = o;
}

__device__ __forceinline__ void store_act_bias2(const v8f& acc,
                                                const float* __restrict__ ba,
                                                const float* __restrict__ bb,
                                                _Float16* act, int ld,
                                                int row0, int col0, int lane) {
  int n  = col0 + (lane & 15);
  int mb = row0 + ((lane >> 4) << 3);
  v8h o;
#pragma unroll
  for (int r = 0; r < 8; r++) {
    o[r] = (_Float16)(acc[r] + ba[mb + r] + bb[mb + r]);
  }
  *(v8h*)(act + (size_t)n * ld + mb) = o;
}

// ------------------------------ Stage 0 ------------------------------------
__global__ void f32_to_f16_kernel(const float* __restrict__ in,
                                  _Float16* __restrict__ out, int n) {
  int i = blockIdx.x * 256 + threadIdx.x;
  if (i < n) out[i] = (_Float16)in[i];
}

// ------------------------------ Stage 1 ------------------------------------
// x1[b][n][h] = sum_c feat16[b][c] * ps_w[c][h][n] + ps_b[h]
// GEMM M=64(b), K=512(c), N=65536(j = h*512+n). 512 WGs x 128 columns.
__global__ __launch_bounds__(256) void pointgen_kernel(
    const float* __restrict__ ps_w, const float* __restrict__ ps_b,
    const _Float16* __restrict__ feat16, _Float16* __restrict__ x1) {
  // ps_w slab double-buffered, TRANSPOSED: wt[buf][j_local][c_local],
  // padded rows (40 halves) for contiguous + bank-spread B-fragment reads.
  __shared__ _Float16 wt[2][128][40];

  const int j0   = blockIdx.x * 128;
  const int t    = threadIdx.x;
  const int lane = t & 31;
  const int wave = t >> 5;

  const int cl = t >> 3;         // 0..31  (c within slab)
  const int jb = (t & 7) * 16;   // 0..112 (j within slab)

  v8f acc[4] = {{}, {}, {}, {}};

  // Prologue: fill slab 0.
  {
    const float* src = ps_w + (size_t)cl * 65536 + j0 + jb;
    float4 v0 = ((const float4*)src)[0];
    float4 v1 = ((const float4*)src)[1];
    float4 v2 = ((const float4*)src)[2];
    float4 v3 = ((const float4*)src)[3];
    float vv[16] = {v0.x, v0.y, v0.z, v0.w, v1.x, v1.y, v1.z, v1.w,
                    v2.x, v2.y, v2.z, v2.w, v3.x, v3.y, v3.z, v3.w};
#pragma unroll
    for (int i = 0; i < 16; i++) wt[0][jb + i][cl] = (_Float16)vv[i];
  }
  __syncthreads();

  for (int kt = 0; kt < 16; kt++) {
    const int k0 = kt * 32;
    const bool have_next = kt < 15;

    // Issue next slab's global loads early (overlap with this slab's WMMAs).
    float vv[16];
    if (have_next) {
      const float* src = ps_w + (size_t)(k0 + 32 + cl) * 65536 + j0 + jb;
      float4 v0 = ((const float4*)src)[0];
      float4 v1 = ((const float4*)src)[1];
      float4 v2 = ((const float4*)src)[2];
      float4 v3 = ((const float4*)src)[3];
      vv[0]=v0.x; vv[1]=v0.y; vv[2]=v0.z; vv[3]=v0.w;
      vv[4]=v1.x; vv[5]=v1.y; vv[6]=v1.z; vv[7]=v1.w;
      vv[8]=v2.x; vv[9]=v2.y; vv[10]=v2.z; vv[11]=v2.w;
      vv[12]=v3.x; vv[13]=v3.y; vv[14]=v3.z; vv[15]=v3.w;
      if (kt < 14) {  // global_prefetch_b8 two slabs ahead
        __builtin_prefetch(ps_w + (size_t)(k0 + 64 + cl) * 65536 + j0 + jb, 0, 1);
      }
    }

    // A-fragments (tiny f16 feat, L2-resident) — issue before B reads.
    v16h afrag[4];
#pragma unroll
    for (int mt = 0; mt < 4; mt++) afrag[mt] = loadA_g(feat16, 512, mt * 16, k0, lane);

    // Compute on current slab.
    v16h bfrag = loadB_lds(&wt[kt & 1][0][0], 40, wave * 16, 0, lane);
#pragma unroll
    for (int mt = 0; mt < 4; mt++) acc[mt] = wmma_f16(afrag[mt], bfrag, acc[mt]);

    // Store next slab (transposed) into the other buffer.
    if (have_next) {
#pragma unroll
      for (int i = 0; i < 16; i++) wt[(kt + 1) & 1][jb + i][cl] = (_Float16)vv[i];
    }
    __syncthreads();
  }

  // Epilogue: j -> (h, n); whole WG shares one h (128 cols within a 512 block).
  const int j = j0 + wave * 16 + (lane & 15);
  const int n = j & 511;
  const int h = j >> 9;
  const float bias = ps_b[h];
#pragma unroll
  for (int mt = 0; mt < 4; mt++) {
    const int mb = mt * 16 + ((lane >> 4) << 3);
#pragma unroll
    for (int r = 0; r < 8; r++) {
      x1[((size_t)(mb + r) * 512 + n) * 128 + h] = (_Float16)(acc[mt][r] + bias);
    }
  }
}

// ------------------------------ Stage 2 ------------------------------------
struct MlpArgs {
  const _Float16 *x1, *feat16;
  const _Float16 *w1_1, *w1_2, *w1_s;
  const _Float16 *w2_1, *w2_2, *w2_s;
  const _Float16 *w3_1, *w3_2, *w3_s;
  const _Float16 *w4_1, *w4_2;
  const float *b1_1, *b1_2, *b1_s;
  const float *b2_1, *b2_2, *b2_s;
  const float *b3_1, *b3_2, *b3_s;
  const float *b4_1, *b4_2;
  float* out;
};

__global__ __launch_bounds__(256) void mlp_fused_kernel(MlpArgs A_) {
  // Activations, point-major. 58.9 KB total static LDS.
  __shared__ _Float16 actIn[32][648];  // [n][ch]: ch 0..127 = x, 128..639 = feat
  __shared__ _Float16 actA[32][136];
  __shared__ _Float16 actB[32][136];

  const int b    = blockIdx.x >> 4;
  const int n0   = (blockIdx.x & 15) * 32;
  const int t    = threadIdx.x;
  const int lane = t & 31;
  const int wave = t >> 5;

  // ---- Fill actIn: x1 part (vector b128 loads thanks to [b][n][h] layout) ----
  {
    int n = t >> 3, c = (t & 7) * 16;
    const v8h* src = (const v8h*)(A_.x1 + ((size_t)b * 512 + n0 + n) * 128 + c);
    *(v8h*)&actIn[n][c]     = src[0];
    *(v8h*)&actIn[n][c + 8] = src[1];
  }
  // ---- Fill actIn: broadcast feat part ----
  {
    int n = t >> 3, c = (t & 7) * 64;
    const v8h* src = (const v8h*)(A_.feat16 + (size_t)b * 512 + c);
    v8h* dst = (v8h*)&actIn[n][128 + c];
#pragma unroll
    for (int i = 0; i < 8; i++) dst[i] = src[i];
  }
  __syncthreads();

  // ---- mlp_1 hidden: relu(w1(128x640) . actIn + b1) -> actA ----
  {
    v8f a0 = {}, a1 = {};
    gemm_acc2(a0, a1, A_.w1_1, 640, wave * 16, &actIn[0][0], 648, 640, lane);
    store_act_relu(a0, A_.b1_1, &actA[0][0], 136, wave * 16, 0,  lane);
    store_act_relu(a1, A_.b1_1, &actA[0][0], 136, wave * 16, 16, lane);
  }
  __syncthreads();

  // ---- mlp_1 out: w2(128x128).actA + ws(128x640).actIn + b2 + bs -> actB ----
  {
    v8f a0 = {}, a1 = {};
    gemm_acc2(a0, a1, A_.w1_2, 128, wave * 16, &actA[0][0], 136, 128, lane);
    gemm_acc2(a0, a1, A_.w1_s, 640, wave * 16, &actIn[0][0], 648, 640, lane);
    store_act_bias2(a0, A_.b1_2, A_.b1_s, &actB[0][0], 136, wave * 16, 0,  lane);
    store_act_bias2(a1, A_.b1_2, A_.b1_s, &actB[0][0], 136, wave * 16, 16, lane);
  }
  __syncthreads();

  // ---- mlp_2 hidden: relu(w1(64x128) . actB + b1) -> actA[.][0..64) ----
  {
    int mt = wave & 3, nt = wave >> 2;
    v8f acc = {};
    gemm_acc(acc, A_.w2_1, 128, mt * 16, &actB[0][0], 136, nt * 16, 128, lane);
    store_act_relu(acc, A_.b2_1, &actA[0][0], 136, mt * 16, nt * 16, lane);
  }
  __syncthreads();

  // ---- mlp_2 out: w2(128x64).actA + ws(128x128).actB + biases -> actIn x-region ----
  {
    v8f a0 = {}, a1 = {};
    gemm_acc2(a0, a1, A_.w2_2, 64, wave * 16, &actA[0][0], 136, 64, lane);
    gemm_acc2(a0, a1, A_.w2_s, 128, wave * 16, &actB[0][0], 136, 128, lane);
    store_act_bias2(a0, A_.b2_2, A_.b2_s, &actIn[0][0], 648, wave * 16, 0,  lane);
    store_act_bias2(a1, A_.b2_2, A_.b2_s, &actIn[0][0], 648, wave * 16, 16, lane);
  }
  __syncthreads();

  // ---- mlp_3 hidden: relu(w1(128x640) . actIn + b1) -> actB ----
  {
    v8f a0 = {}, a1 = {};
    gemm_acc2(a0, a1, A_.w3_1, 640, wave * 16, &actIn[0][0], 648, 640, lane);
    store_act_relu(a0, A_.b3_1, &actB[0][0], 136, wave * 16, 0,  lane);
    store_act_relu(a1, A_.b3_1, &actB[0][0], 136, wave * 16, 16, lane);
  }
  __syncthreads();

  // ---- mlp_3 out: w2(128x128).actB + ws(128x640).actIn + biases -> actA ----
  {
    v8f a0 = {}, a1 = {};
    gemm_acc2(a0, a1, A_.w3_2, 128, wave * 16, &actB[0][0], 136, 128, lane);
    gemm_acc2(a0, a1, A_.w3_s, 640, wave * 16, &actIn[0][0], 648, 640, lane);
    store_act_bias2(a0, A_.b3_2, A_.b3_s, &actA[0][0], 136, wave * 16, 0,  lane);
    store_act_bias2(a1, A_.b3_2, A_.b3_s, &actA[0][0], 136, wave * 16, 16, lane);
  }
  __syncthreads();

  // ---- mlp_4 hidden: relu(w1(64x128) . actA + b1) -> actB[.][0..64) ----
  {
    int mt = wave & 3, nt = wave >> 2;
    v8f acc = {};
    gemm_acc(acc, A_.w4_1, 128, mt * 16, &actA[0][0], 136, nt * 16, 128, lane);
    store_act_relu(acc, A_.b4_1, &actB[0][0], 136, mt * 16, nt * 16, lane);
  }
  __syncthreads();

  // ---- mlp_4 out: w2(3x64) . actB + b2 -> out (B,3,512) fp32 ----
  if (wave < 2) {
    const int nt = wave;
    v8f acc = {};
    for (int k0 = 0; k0 < 64; k0 += 32) {
      v16h a = loadA_g_guard(A_.w4_2, 64, 3, 0, k0, lane);
      v16h bb = loadB_lds(&actB[0][0], 136, nt * 16, k0, lane);
      acc = wmma_f16(a, bb, acc);
    }
    const int n  = n0 + nt * 16 + (lane & 15);
    const int mb = (lane >> 4) << 3;
#pragma unroll
    for (int r = 0; r < 8; r++) {
      int m = mb + r;
      if (m < 3) A_.out[((size_t)b * 3 + m) * 512 + n] = acc[r] + A_.b4_2[m];
    }
  }
}

// ------------------------------ Launch -------------------------------------
extern "C" void kernel_launch(void* const* d_in, const int* in_sizes, int n_in,
                              void* d_out, int out_size, void* d_ws, size_t ws_size,
                              hipStream_t stream) {
  (void)in_sizes; (void)n_in; (void)out_size; (void)ws_size;

  // Inputs (setup_inputs dict order: feat, then params in insertion order).
  const float* ps_w  = (const float*)d_in[1];
  const float* ps_b  = (const float*)d_in[2];

  // Workspace layout (halves), all chunks 16B-aligned.
  _Float16* ws16    = (_Float16*)d_ws;
  _Float16* feat16  = ws16;                       // 32768
  _Float16* x1      = ws16 + 32768;               // 4194304  ([b][n][h])
  _Float16* Wbase   = ws16 + 32768 + 4194304;
  _Float16* w1_1 = Wbase;            // 81920 (128x640)
  _Float16* w1_2 = Wbase + 81920;    // 16384 (128x128)
  _Float16* w1_s = Wbase + 98304;    // 81920 (128x640)
  _Float16* w2_1 = Wbase + 180224;   // 8192  (64x128)
  _Float16* w2_2 = Wbase + 188416;   // 8192  (128x64)
  _Float16* w2_s = Wbase + 196608;   // 16384 (128x128)
  _Float16* w3_1 = Wbase + 212992;   // 81920
  _Float16* w3_2 = Wbase + 294912;   // 16384
  _Float16* w3_s = Wbase + 311296;   // 81920
  _Float16* w4_1 = Wbase + 393216;   // 8192  (64x128)
  _Float16* w4_2 = Wbase + 401408;   // 192   (3x64)

  struct Cvt { int src; _Float16* dst; int n; };
  const Cvt cv[12] = {
      {0,  feat16, 32768}, {3,  w1_1, 81920}, {5,  w1_2, 16384}, {7,  w1_s, 81920},
      {9,  w2_1, 8192},    {11, w2_2, 8192},  {13, w2_s, 16384}, {15, w3_1, 81920},
      {17, w3_2, 16384},   {19, w3_s, 81920}, {21, w4_1, 8192},  {23, w4_2, 192}};
  for (int i = 0; i < 12; i++) {
    int blocks = (cv[i].n + 255) / 256;
    f32_to_f16_kernel<<<blocks, 256, 0, stream>>>((const float*)d_in[cv[i].src],
                                                  cv[i].dst, cv[i].n);
  }

  // Stage 1: seed GEMM. 512 WGs x 128 columns of the 65536-wide output.
  pointgen_kernel<<<512, 256, 0, stream>>>(ps_w, ps_b, feat16, x1);

  // Stage 2: fused MLP stack. 64 batches x 16 point-tiles.
  MlpArgs a;
  a.x1 = x1; a.feat16 = feat16;
  a.w1_1 = w1_1; a.w1_2 = w1_2; a.w1_s = w1_s;
  a.w2_1 = w2_1; a.w2_2 = w2_2; a.w2_s = w2_s;
  a.w3_1 = w3_1; a.w3_2 = w3_2; a.w3_s = w3_s;
  a.w4_1 = w4_1; a.w4_2 = w4_2;
  a.b1_1 = (const float*)d_in[4];  a.b1_2 = (const float*)d_in[6];  a.b1_s = (const float*)d_in[8];
  a.b2_1 = (const float*)d_in[10]; a.b2_2 = (const float*)d_in[12]; a.b2_s = (const float*)d_in[14];
  a.b3_1 = (const float*)d_in[16]; a.b3_2 = (const float*)d_in[18]; a.b3_s = (const float*)d_in[20];
  a.b4_1 = (const float*)d_in[22]; a.b4_2 = (const float*)d_in[24];
  a.out = (float*)d_out;
  mlp_fused_kernel<<<64 * 16, 256, 0, stream>>>(a);
}
